// GnBlock_45509473468803
// MI455X (gfx1250) — matrix operand order repaired and checked by reading, not compiled, not run on hardware
//
#include <hip/hip_runtime.h>
#include <hip/hip_bf16.h>

// ---------------------------------------------------------------------------
// CDNA5 / gfx1250 GraphNet block: edge MLP+LN (+scatter-add) then node MLP+LN.
// fp32 I/O, bf16 WMMA (v_wmma_f32_16x16x32_bf16) with fp32 accumulation.
// ---------------------------------------------------------------------------

typedef __attribute__((ext_vector_type(16))) __bf16 v16bf;
typedef __attribute__((ext_vector_type(8)))  __bf16 v8bf;
typedef __attribute__((ext_vector_type(8)))  float  v8f;

#define HID 128

// Swizzled-weight offsets in workspace (units: bf16 halves)
#define EB_W0_OFF 0
#define EB_W1_OFF 49152
#define EB_W2_OFF 65536
#define EB_W3_OFF 81920
#define NB_W0_OFF 98304
#define NB_W1_OFF 131072
#define NB_W2_OFF 147456
#define NB_W3_OFF 163840

// Padded LDS row strides (halves): +8 keeps 16 b128 lane-reads conflict-free
#define A_STRIDE_EDGE 392   // 384 cols
#define A_STRIDE_NODE 264   // 256 cols
#define H_STRIDE      136   // 128 cols

// ---------------------------------------------------------------------------
// Weight swizzle: fp32 [K x 128] row-major -> bf16 fragment-major so a lane's
// entire B-fragment (v16bf, 32 bytes) is one contiguous global load.
// B-frag layout (16-bit B, 32x16): lane L -> n = L%16;
//   element s holds k = k0 + (L>=16 ? 8 : 0) + (s & 7) + (s >= 8 ? 16 : 0).
// Linear dst index: (((ks*8 + t)*32 + L)*16 + j2)
// ---------------------------------------------------------------------------
__global__ void swizzle_w_kernel(const float* __restrict__ W,
                                 __bf16* __restrict__ dst, int K) {
  int i = blockIdx.x * blockDim.x + threadIdx.x;
  int total = K * HID;
  if (i >= total) return;
  int j2   = i & 15;
  int lane = (i >> 4) & 31;
  int frag = i >> 9;
  int t    = frag & 7;
  int ks   = frag >> 3;
  int k = 32 * ks + ((lane >= 16) ? 8 : 0) + (j2 & 7) + ((j2 & 8) ? 16 : 0);
  int n = 16 * t + (lane & 15);
  dst[i] = (__bf16)W[(size_t)k * HID + n];
}

__global__ void zero_kernel(float* __restrict__ p, long long n) {
  long long i = (long long)blockIdx.x * blockDim.x + threadIdx.x;
  if (i < n) p[i] = 0.0f;
}

// ---------------------------------------------------------------------------
// Wave-level GEMM: acc[16 rows x 128 cols] = A(16 x KDIM, bf16 in LDS) @ W.
// A-frag (16-bit A, 16x32): lanes 0-15 -> m=lane, K={k0..k0+7, k0+16..k0+23};
// lanes 16-31 -> m=lane-16, K shifted by 8. Two ds_load_b128 per fragment.
// ---------------------------------------------------------------------------
template <int KDIM, int STRIDE>
__device__ __forceinline__ void wave_layer(const __bf16* __restrict__ Wz,
                                           const __bf16* __restrict__ A,
                                           int lane, v8f acc[8]) {
  const __bf16* rowp = A + (lane & 15) * STRIDE;
  const int khalf = (lane >= 16) ? 8 : 0;
  v8f z = {};
#pragma unroll
  for (int t = 0; t < 8; ++t) acc[t] = z;
#pragma unroll
  for (int ks = 0; ks < KDIM / 32; ++ks) {
    const __bf16* p = rowp + ks * 32 + khalf;
    v8bf lo = *(const v8bf*)p;
    v8bf hi = *(const v8bf*)(p + 16);
    v16bf a = __builtin_shufflevector(lo, hi, 0, 1, 2, 3, 4, 5, 6, 7,
                                      8, 9, 10, 11, 12, 13, 14, 15);
#pragma unroll
    for (int t = 0; t < 8; ++t) {
      v16bf b = *(const v16bf*)(Wz + ((size_t)(ks * 8 + t) * 32 + lane) * 16);
      acc[t] = __builtin_amdgcn_wmma_f32_16x16x32_bf16(
          false, a, false, b, (short)0, acc[t], false, false);
    }
  }
}

// bias + ReLU, write bf16 H tile back to (wave-private) LDS for the next layer
__device__ __forceinline__ void bias_relu_store(v8f acc[8],
                                                const float* __restrict__ bias,
                                                __bf16* __restrict__ H,
                                                int lane) {
  const int nlo = lane & 15;
  const int mh = (lane >= 16) ? 8 : 0;
#pragma unroll
  for (int t = 0; t < 8; ++t) {
    float bv = bias[t * 16 + nlo];
#pragma unroll
    for (int j = 0; j < 8; ++j) {
      float v = acc[t][j] + bv;
      H[(j + mh) * H_STRIDE + (t * 16 + nlo)] = (__bf16)(v > 0.0f ? v : 0.0f);
    }
  }
}

// Sum across the 16-lane half this lane belongs to (masks stay inside halves)
__device__ __forceinline__ float half16_sum(float v) {
  v += __shfl_xor(v, 1, 32);
  v += __shfl_xor(v, 2, 32);
  v += __shfl_xor(v, 4, 32);
  v += __shfl_xor(v, 8, 32);
  return v;
}

// ---------------------------------------------------------------------------
// Edge block: gather [x_node[src] | x_node[dst] | x_edge] -> MLP -> LN ->
// residual edge output + atomic scatter-add into agg (node region of d_out).
// 4 waves / block, 16 edges per wave, wave-private LDS tile (no barriers).
// ---------------------------------------------------------------------------
__global__ __launch_bounds__(128) void edge_block_kernel(
    const float* __restrict__ x_node, const float* __restrict__ x_edge,
    const int* __restrict__ src, const int* __restrict__ dst,
    const __bf16* __restrict__ Wz, const float* __restrict__ b0,
    const float* __restrict__ b1, const float* __restrict__ b2,
    const float* __restrict__ b3, const float* __restrict__ ln_g,
    const float* __restrict__ ln_b, float* __restrict__ out_edge,
    float* __restrict__ agg, int E) {
  extern __shared__ __bf16 smem[];
  const int wave = threadIdx.x >> 5;
  const int lane = threadIdx.x & 31;
  __bf16* A = smem + wave * (16 * A_STRIDE_EDGE);
  const int row0 = blockIdx.x * 64 + wave * 16;

  // ---- build A tile: 16 rows x 384 cols bf16 (float4-granular gather) ----
  for (int it = lane; it < 16 * 96; it += 32) {
    int m = it / 96;
    int q = it - m * 96;
    int e = row0 + m;
    if (e < E) {
      int col4 = q * 4;
      const float* sp;
      if (q < 32)
        sp = x_node + (size_t)src[e] * HID + col4;
      else if (q < 64)
        sp = x_node + (size_t)dst[e] * HID + (col4 - 128);
      else
        sp = x_edge + (size_t)e * HID + (col4 - 256);
      float4 v = *(const float4*)sp;
      __bf16* d = A + m * A_STRIDE_EDGE + col4;
      d[0] = (__bf16)v.x; d[1] = (__bf16)v.y;
      d[2] = (__bf16)v.z; d[3] = (__bf16)v.w;
    }
  }

  // ---- MLP: 384->128 relu, 128->128 relu x2, 128->128 ----
  v8f acc[8];
  wave_layer<384, A_STRIDE_EDGE>(Wz + EB_W0_OFF, A, lane, acc);
  bias_relu_store(acc, b0, A, lane);
  wave_layer<128, H_STRIDE>(Wz + EB_W1_OFF, A, lane, acc);
  bias_relu_store(acc, b1, A, lane);
  wave_layer<128, H_STRIDE>(Wz + EB_W2_OFF, A, lane, acc);
  bias_relu_store(acc, b2, A, lane);
  wave_layer<128, H_STRIDE>(Wz + EB_W3_OFF, A, lane, acc);

  // ---- bias + LayerNorm + residual + scatter-add ----
  const int nlo = lane & 15;
  const int mh = (lane >= 16) ? 8 : 0;
  float b3v[8], gv[8], btv[8];
#pragma unroll
  for (int t = 0; t < 8; ++t) {
    int n = t * 16 + nlo;
    b3v[t] = b3[n]; gv[t] = ln_g[n]; btv[t] = ln_b[n];
  }
#pragma unroll
  for (int j = 0; j < 8; ++j) {
    float s = 0.0f, sq = 0.0f;
#pragma unroll
    for (int t = 0; t < 8; ++t) {
      float v = acc[t][j] + b3v[t];
      acc[t][j] = v;
      s += v; sq += v * v;
    }
    s = half16_sum(s);
    sq = half16_sum(sq);
    float mu = s * (1.0f / 128.0f);
    float var = sq * (1.0f / 128.0f) - mu * mu;
    float rs = rsqrtf(var + 1e-5f);
    int e = row0 + j + mh;
    if (e < E) {
      int se = src[e], de = dst[e];
      size_t eb = (size_t)e * HID;
      size_t sb = (size_t)se * HID, db = (size_t)de * HID;
#pragma unroll
      for (int t = 0; t < 8; ++t) {
        int n = t * 16 + nlo;
        float y = (acc[t][j] - mu) * rs * gv[t] + btv[t];
        out_edge[eb + n] = x_edge[eb + n] + y;
        atomicAdd(&agg[sb + n], y);
        atomicAdd(&agg[db + n], y);
      }
    }
  }
}

// ---------------------------------------------------------------------------
// Node block: [x_node | agg] -> MLP -> LN -> residual node output.
// agg lives in the node region of d_out; each block reads only its own rows
// before overwriting them with the final output.
// ---------------------------------------------------------------------------
__global__ __launch_bounds__(128) void node_block_kernel(
    const float* __restrict__ x_node, const __bf16* __restrict__ Wz,
    const float* __restrict__ b0, const float* __restrict__ b1,
    const float* __restrict__ b2, const float* __restrict__ b3,
    const float* __restrict__ ln_g, const float* __restrict__ ln_b,
    float* __restrict__ out_node /* == agg */, int N) {
  extern __shared__ __bf16 smem[];
  const int wave = threadIdx.x >> 5;
  const int lane = threadIdx.x & 31;
  __bf16* A = smem + wave * (16 * A_STRIDE_NODE);
  const int row0 = blockIdx.x * 64 + wave * 16;

  for (int it = lane; it < 16 * 64; it += 32) {
    int m = it / 64;
    int q = it - m * 64;
    int v = row0 + m;
    if (v < N) {
      int col4 = q * 4;
      const float* sp = (q < 32) ? x_node + (size_t)v * HID + col4
                                 : out_node + (size_t)v * HID + (col4 - 128);
      float4 x = *(const float4*)sp;
      __bf16* d = A + m * A_STRIDE_NODE + col4;
      d[0] = (__bf16)x.x; d[1] = (__bf16)x.y;
      d[2] = (__bf16)x.z; d[3] = (__bf16)x.w;
    }
  }

  v8f acc[8];
  wave_layer<256, A_STRIDE_NODE>(Wz + NB_W0_OFF, A, lane, acc);
  bias_relu_store(acc, b0, A, lane);
  wave_layer<128, H_STRIDE>(Wz + NB_W1_OFF, A, lane, acc);
  bias_relu_store(acc, b1, A, lane);
  wave_layer<128, H_STRIDE>(Wz + NB_W2_OFF, A, lane, acc);
  bias_relu_store(acc, b2, A, lane);
  wave_layer<128, H_STRIDE>(Wz + NB_W3_OFF, A, lane, acc);

  const int nlo = lane & 15;
  const int mh = (lane >= 16) ? 8 : 0;
  float b3v[8], gv[8], btv[8];
#pragma unroll
  for (int t = 0; t < 8; ++t) {
    int n = t * 16 + nlo;
    b3v[t] = b3[n]; gv[t] = ln_g[n]; btv[t] = ln_b[n];
  }
#pragma unroll
  for (int j = 0; j < 8; ++j) {
    float s = 0.0f, sq = 0.0f;
#pragma unroll
    for (int t = 0; t < 8; ++t) {
      float v = acc[t][j] + b3v[t];
      acc[t][j] = v;
      s += v; sq += v * v;
    }
    s = half16_sum(s);
    sq = half16_sum(sq);
    float mu = s * (1.0f / 128.0f);
    float var = sq * (1.0f / 128.0f) - mu * mu;
    float rs = rsqrtf(var + 1e-5f);
    int v = row0 + j + mh;
    if (v < N) {
      size_t vb = (size_t)v * HID;
#pragma unroll
      for (int t = 0; t < 8; ++t) {
        int n = t * 16 + nlo;
        float y = (acc[t][j] - mu) * rs * gv[t] + btv[t];
        out_node[vb + n] = x_node[vb + n] + y;
      }
    }
  }
}

// ---------------------------------------------------------------------------
extern "C" void kernel_launch(void* const* d_in, const int* in_sizes, int n_in,
                              void* d_out, int out_size, void* d_ws,
                              size_t ws_size, hipStream_t stream) {
  const float* x_node = (const float*)d_in[0];
  const float* x_edge = (const float*)d_in[1];
  const int* eidx = (const int*)d_in[2];
  // params: eb {W0,b0,W1,b1,W2,b2,W3,b3,ln_g,ln_b}, then nb {...}
  const float* ebW0 = (const float*)d_in[3];
  const float* ebb0 = (const float*)d_in[4];
  const float* ebW1 = (const float*)d_in[5];
  const float* ebb1 = (const float*)d_in[6];
  const float* ebW2 = (const float*)d_in[7];
  const float* ebb2 = (const float*)d_in[8];
  const float* ebW3 = (const float*)d_in[9];
  const float* ebb3 = (const float*)d_in[10];
  const float* ebg = (const float*)d_in[11];
  const float* ebb = (const float*)d_in[12];
  const float* nbW0 = (const float*)d_in[13];
  const float* nbb0 = (const float*)d_in[14];
  const float* nbW1 = (const float*)d_in[15];
  const float* nbb1 = (const float*)d_in[16];
  const float* nbW2 = (const float*)d_in[17];
  const float* nbb2 = (const float*)d_in[18];
  const float* nbW3 = (const float*)d_in[19];
  const float* nbb3 = (const float*)d_in[20];
  const float* nbg = (const float*)d_in[21];
  const float* nbb = (const float*)d_in[22];

  const int N = in_sizes[0] / HID;
  const int E = in_sizes[1] / HID;
  const int* src = eidx;
  const int* dst = eidx + E;

  float* out_node = (float*)d_out;                  // also the agg accumulator
  float* out_edge = out_node + (size_t)N * HID;
  __bf16* wz = (__bf16*)d_ws;

  // 1) swizzle all weights fp32 -> bf16 fragment-major
  auto swz = [&](const float* W, int K, size_t off) {
    int total = K * HID;
    swizzle_w_kernel<<<(total + 255) / 256, 256, 0, stream>>>(W, wz + off, K);
  };
  swz(ebW0, 384, EB_W0_OFF);
  swz(ebW1, 128, EB_W1_OFF);
  swz(ebW2, 128, EB_W2_OFF);
  swz(ebW3, 128, EB_W3_OFF);
  swz(nbW0, 256, NB_W0_OFF);
  swz(nbW1, 128, NB_W1_OFF);
  swz(nbW2, 128, NB_W2_OFF);
  swz(nbW3, 128, NB_W3_OFF);

  // 2) zero the aggregation buffer (node region of d_out)
  long long nz = (long long)N * HID;
  zero_kernel<<<(int)((nz + 255) / 256), 256, 0, stream>>>(out_node, nz);

  // 3) edge block (+ scatter-add into agg)
  int eblocks = (E + 63) / 64;
  size_t smem_edge = (size_t)4 * 16 * A_STRIDE_EDGE * sizeof(__bf16);
  edge_block_kernel<<<eblocks, 128, smem_edge, stream>>>(
      x_node, x_edge, src, dst, wz, ebb0, ebb1, ebb2, ebb3, ebg, ebb,
      out_edge, out_node, E);

  // 4) node block (reads agg, overwrites with final node output)
  int nblocks = (N + 63) / 64;
  size_t smem_node = (size_t)4 * 16 * A_STRIDE_NODE * sizeof(__bf16);
  node_block_kernel<<<nblocks, 128, smem_node, stream>>>(
      x_node, wz, nbb0, nbb1, nbb2, nbb3, nbg, nbb, out_node, N);
}
